// CoAttentionLayer_57140244906000
// MI455X (gfx1250) — matrix operand order repaired and checked by reading, not compiled, not run on hardware
//
#include <hip/hip_runtime.h>
#include <hip/hip_bf16.h>
#include <math.h>

// Problem constants (match reference)
constexpr int BATCH = 32;
constexpr int LC = 512;
constexpr int LP = 1024;
constexpr int DD = 1024;
constexpr float NEGV = -1e9f;

typedef __attribute__((ext_vector_type(16))) __bf16 v16bf;
typedef __attribute__((ext_vector_type(8)))  __bf16 v8bf;
typedef __attribute__((ext_vector_type(4)))  __bf16 v4bf;
typedef __attribute__((ext_vector_type(8)))  float  v8f;
typedef __attribute__((ext_vector_type(4)))  float  v4f;

constexpr int BM = 128, BN = 128, BK = 32;

__device__ inline v4f  loadv(const float* p)  { return *(const v4f*)p; }
__device__ inline v4bf loadv(const __bf16* p) { return *(const v4bf*)p; }
__device__ inline v4bf to_bf(v4f v)  { return __builtin_convertvector(v, v4bf); }
__device__ inline v4bf to_bf(v4bf v) { return v; }

template <typename T> struct VecOf            { using type = v4bf; };
template <>           struct VecOf<float>     { using type = v4f;  };

// A-fragment (16x32 bf16, M = lane&15, K-halves split by lane>>4):
//   elems 0..7  = K in [8*h, 8*h+8);  elems 8..15 = K in [16+8*h, 16+8*h+8)
__device__ inline v16bf load_frag_a(const __bf16* s, int rowBase, int lane) {
  int m  = rowBase + (lane & 15);
  int kh = (lane >> 4) << 3;  // 0 or 8
  union { v16bf f; v8bf h[2]; } u;
  u.h[0] = *(const v8bf*)(s + m * BK + kh);
  u.h[1] = *(const v8bf*)(s + m * BK + 16 + kh);
  return u.f;
}
// B-fragment (32x16 bf16, N = lane&15): elems 0..7 = K in [8*h, 8*h+8); elems 8..15 = K in [16+8*h, 16+8*h+8),
// read from LDS tile stored transposed as [N][K].
__device__ inline v16bf load_frag_b(const __bf16* s, int rowBase, int lane) {
  int n  = rowBase + (lane & 15);
  int kh = (lane >> 4) << 3;  // 0 or 8
  union { v16bf f; v8bf h[2]; } u;
  u.h[0] = *(const v8bf*)(s + n * BK + kh);
  u.h[1] = *(const v8bf*)(s + n * BK + 16 + kh);
  return u.f;
}

// Double-buffered tiled bf16 WMMA GEMM: C = A * op(B), f32 accumulate.
//   B_KN = false: B stored [N][K] ("NT");  true: B stored [K][N] ("NN")
//   EPI  = 0: store f32 C;  EPI = 1: store bf16 tanh(C + bias[n])
template <typename TA, typename TB, bool B_KN, int EPI>
__global__ __launch_bounds__(256) void gemm_wmma_kernel(
    const TA* __restrict__ Ag, long long strideA, int lda,
    const TB* __restrict__ Bg, long long strideB, int ldb,
    void* __restrict__ Cg, long long strideC, int ldc,
    const float* __restrict__ bias, int K)
{
  __shared__ __align__(16) __bf16 sA[2][BM * BK];
  __shared__ __align__(16) __bf16 sB[2][BN * BK];  // stored [n][k]

  const int tid  = threadIdx.x;
  const int lane = tid & 31;
  const int wid  = tid >> 5;   // 8 waves
  const int wm   = wid >> 2;   // 0..1  (64 rows)
  const int wn   = wid & 3;    // 0..3  (32 cols)
  const int m0   = blockIdx.y * BM;
  const int n0   = blockIdx.x * BN;
  const int bz   = blockIdx.z;

  const TA* A  = Ag + (long long)bz * strideA;
  const TB* Bm = Bg + (long long)bz * strideB;

  using AV = typename VecOf<TA>::type;
  using BV = typename VecOf<TB>::type;
  AV ra[4];
  BV rb[4];

  // Per-thread staging pointers (induction; bumped by BK each k-step)
  const int rowA = tid >> 3;            // 0..31, +32*i
  const int colA = (tid & 7) << 2;      // 0..28
  const int krB  = tid >> 5;            // NN: 0..7, +8*i
  const int ncB  = (tid & 31) << 2;     // NN: 0..124
  const TA* aptr[4];
  const TB* bptr[4];
#pragma unroll
  for (int i = 0; i < 4; ++i)
    aptr[i] = A + (long long)(m0 + rowA + 32 * i) * lda + colA;
#pragma unroll
  for (int i = 0; i < 4; ++i) {
    if (!B_KN) bptr[i] = Bm + (long long)(n0 + rowA + 32 * i) * ldb + colA;
    else       bptr[i] = Bm + (long long)(krB + 8 * i) * ldb + (n0 + ncB);
  }

  auto fetch = [&]() {
#pragma unroll
    for (int i = 0; i < 4; ++i) { ra[i] = loadv(aptr[i]); aptr[i] += BK; }
#pragma unroll
    for (int i = 0; i < 4; ++i) {
      rb[i] = loadv(bptr[i]);
      if (!B_KN) bptr[i] += BK;
      else       bptr[i] += (long long)BK * ldb;
    }
  };
  auto stash = [&](int tb) {
    __bf16* da = &sA[tb][0];
    __bf16* db = &sB[tb][0];
#pragma unroll
    for (int i = 0; i < 4; ++i)
      *(v4bf*)(da + tid * 4 + 1024 * i) = to_bf(ra[i]);   // tid*4 == rowA*BK+colA
    if (!B_KN) {
#pragma unroll
      for (int i = 0; i < 4; ++i)
        *(v4bf*)(db + tid * 4 + 1024 * i) = to_bf(rb[i]);
    } else {
#pragma unroll
      for (int i = 0; i < 4; ++i) {                       // LDS scatter-transpose
        v4bf v = to_bf(rb[i]);
        int kk = krB + 8 * i;
        db[(ncB + 0) * BK + kk] = v.x;
        db[(ncB + 1) * BK + kk] = v.y;
        db[(ncB + 2) * BK + kk] = v.z;
        db[(ncB + 3) * BK + kk] = v.w;
      }
    }
  };

  v8f acc[4][2];
#pragma unroll
  for (int i = 0; i < 4; ++i)
#pragma unroll
    for (int j = 0; j < 2; ++j)
#pragma unroll
      for (int e = 0; e < 8; ++e) acc[i][j][e] = 0.f;

  const int nk = K / BK;
  fetch();
  stash(0);
  __syncthreads();

  int buf = 0;
  for (int kt = 0; kt < nk; ++kt) {
    const bool more = (kt + 1 < nk);
    if (more) fetch();                 // next tile's global loads in flight...

    { // ...while this tile's WMMAs execute
      const __bf16* pa = &sA[buf][0];
      const __bf16* pb = &sB[buf][0];
      v16bf afrag[4], bfrag[2];
#pragma unroll
      for (int mi = 0; mi < 4; ++mi) afrag[mi] = load_frag_a(pa, wm * 64 + mi * 16, lane);
#pragma unroll
      for (int ni = 0; ni < 2; ++ni) bfrag[ni] = load_frag_b(pb, wn * 32 + ni * 16, lane);
#pragma unroll
      for (int mi = 0; mi < 4; ++mi)
#pragma unroll
        for (int ni = 0; ni < 2; ++ni)
          acc[mi][ni] = __builtin_amdgcn_wmma_f32_16x16x32_bf16(
              false, afrag[mi], false, bfrag[ni], (short)0, acc[mi][ni], false, false);
    }

    if (more) stash(buf ^ 1);
    __syncthreads();
    buf ^= 1;
  }

  // ---- epilogue: C/D layout = VGPR g -> rows g (lanes 0-15) / g+8 (lanes 16-31) ----
  const int hl = lane >> 4;
  const int nl = lane & 15;
#pragma unroll
  for (int mi = 0; mi < 4; ++mi) {
#pragma unroll
    for (int ni = 0; ni < 2; ++ni) {
      int colg = n0 + wn * 32 + ni * 16 + nl;
      float bv = (EPI == 1) ? bias[colg] : 0.f;
#pragma unroll
      for (int g = 0; g < 8; ++g) {
        int rowg = m0 + wm * 64 + mi * 16 + g + 8 * hl;
        float cv = acc[mi][ni][g];
        if (EPI == 1) {
          __bf16* out = (__bf16*)Cg + (long long)bz * strideC;
          out[(long long)rowg * ldc + colg] = (__bf16)tanhf(cv + bv);
        } else {
          float* out = (float*)Cg + (long long)bz * strideC;
          out[(long long)rowg * ldc + colg] = cv;
        }
      }
    }
  }
}

// ---------------- softmax helpers (wave32) ----------------
__device__ inline float wave_max(float v) {
#pragma unroll
  for (int o = 16; o > 0; o >>= 1) v = fmaxf(v, __shfl_xor(v, o, 32));
  return v;
}
__device__ inline float wave_sum(float v) {
#pragma unroll
  for (int o = 16; o > 0; o >>= 1) v += __shfl_xor(v, o, 32);
  return v;
}

// comp_attention: in-place row softmax over LP, mask -> -1e9
__global__ __launch_bounds__(256) void comp_softmax_kernel(
    float* __restrict__ att, const float* __restrict__ cmask,
    const float* __restrict__ pmask)
{
  __shared__ float redm[8], reds[8];
  const int b = blockIdx.y, c = blockIdx.x, tid = threadIdx.x;
  float* row = att + ((long long)b * LC + c) * LP;
  const float rm = cmask[b * LC + c];

  float v[4];
  float mx = -3.402823466e38f;
#pragma unroll
  for (int i = 0; i < 4; ++i) {
    int p = tid + 256 * i;
    float a = row[p];
    a = (rm * pmask[b * LP + p] == 0.f) ? NEGV : a;
    v[i] = a;
    mx = fmaxf(mx, a);
  }
  mx = wave_max(mx);
  if ((tid & 31) == 0) redm[tid >> 5] = mx;
  __syncthreads();
#pragma unroll
  for (int i = 0; i < 8; ++i) mx = fmaxf(mx, redm[i]);

  float s = 0.f;
#pragma unroll
  for (int i = 0; i < 4; ++i) { v[i] = __expf(v[i] - mx); s += v[i]; }
  s = wave_sum(s);
  if ((tid & 31) == 0) reds[tid >> 5] = s;
  __syncthreads();
  s = 0.f;
#pragma unroll
  for (int i = 0; i < 8; ++i) s += reds[i];
  const float inv = 1.f / s;
#pragma unroll
  for (int i = 0; i < 4; ++i) row[tid + 256 * i] = v[i] * inv;
}

// prot_attention[b,p,c] = softmax_c(aff[b,c,p]) with mask
__global__ __launch_bounds__(256) void prot_softmax_kernel(
    const float* __restrict__ aff, float* __restrict__ pat,
    const float* __restrict__ cmask, const float* __restrict__ pmask)
{
  __shared__ float redm[8], reds[8];
  const int b = blockIdx.y, p = blockIdx.x, tid = threadIdx.x;
  const float* colbase = aff + (long long)b * LC * LP + p;
  const float pm = pmask[b * LP + p];

  float v[2];
  float mx = -3.402823466e38f;
#pragma unroll
  for (int i = 0; i < 2; ++i) {
    int c = tid + 256 * i;
    float a = colbase[(long long)c * LP];
    a = (pm * cmask[b * LC + c] == 0.f) ? NEGV : a;
    v[i] = a;
    mx = fmaxf(mx, a);
  }
  mx = wave_max(mx);
  if ((tid & 31) == 0) redm[tid >> 5] = mx;
  __syncthreads();
#pragma unroll
  for (int i = 0; i < 8; ++i) mx = fmaxf(mx, redm[i]);

  float s = 0.f;
#pragma unroll
  for (int i = 0; i < 2; ++i) { v[i] = __expf(v[i] - mx); s += v[i]; }
  s = wave_sum(s);
  if ((tid & 31) == 0) reds[tid >> 5] = s;
  __syncthreads();
  s = 0.f;
#pragma unroll
  for (int i = 0; i < 8; ++i) s += reds[i];
  const float inv = 1.f / s;

  float* orow = pat + ((long long)b * LP + p) * LC;
#pragma unroll
  for (int i = 0; i < 2; ++i) orow[tid + 256 * i] = v[i] * inv;
}

extern "C" void kernel_launch(void* const* d_in, const int* in_sizes, int n_in,
                              void* d_out, int out_size, void* d_ws, size_t ws_size,
                              hipStream_t stream) {
  (void)in_sizes; (void)n_in; (void)out_size; (void)ws_size;
  const float* comp_feat = (const float*)d_in[0];  // [B,LC,D]
  const float* prot_feat = (const float*)d_in[1];  // [B,LP,D]
  const float* comp_mask = (const float*)d_in[2];  // [B,LC]
  const float* prot_mask = (const float*)d_in[3];  // [B,LP]
  const float* W_comp    = (const float*)d_in[4];  // [D,D]
  const float* b_comp    = (const float*)d_in[5];  // [D]
  const float* W_prot    = (const float*)d_in[6];  // [D,D]
  const float* b_prot    = (const float*)d_in[7];  // [D]

  float* out = (float*)d_out;
  float* comp_attended = out;                                           // [B,LC,D]
  float* prot_attended = comp_attended + (size_t)BATCH * LC * DD;       // [B,LP,D]
  float* comp_attn     = prot_attended + (size_t)BATCH * LP * DD;       // [B,LC,LP]
  float* prot_attn     = comp_attn + (size_t)BATCH * LC * LP;           // [B,LP,LC]

  __bf16* comp_trans = (__bf16*)d_ws;                                   // [B*LC, D] bf16
  __bf16* prot_trans = comp_trans + (size_t)BATCH * LC * DD;            // [B*LP, D] bf16

  const dim3 blk(256);

  // 1-2) comp_trans / prot_trans = tanh(X @ W^T + b) -> bf16 (NT, fused epilogue)
  gemm_wmma_kernel<float, float, false, 1>
      <<<dim3(DD / BN, (BATCH * LC) / BM, 1), blk, 0, stream>>>(
          comp_feat, 0, DD, W_comp, 0, DD, (void*)comp_trans, 0, DD, b_comp, DD);
  gemm_wmma_kernel<float, float, false, 1>
      <<<dim3(DD / BN, (BATCH * LP) / BM, 1), blk, 0, stream>>>(
          prot_feat, 0, DD, W_prot, 0, DD, (void*)prot_trans, 0, DD, b_prot, DD);

  // 3) raw affinity = comp_trans @ prot_trans^T -> f32, staged into comp_attn region
  gemm_wmma_kernel<__bf16, __bf16, false, 0>
      <<<dim3(LP / BN, LC / BM, BATCH), blk, 0, stream>>>(
          comp_trans, (long long)LC * DD, DD,
          prot_trans, (long long)LP * DD, DD,
          (void*)comp_attn, (long long)LC * LP, LP, nullptr, DD);

  // 4) prot_attention = softmax over LC of affinity^T (reads raw affinity first)
  prot_softmax_kernel<<<dim3(LP, BATCH), blk, 0, stream>>>(
      comp_attn, prot_attn, comp_mask, prot_mask);

  // 5) comp_attention = row softmax over LP, in place
  comp_softmax_kernel<<<dim3(LC, BATCH), blk, 0, stream>>>(
      comp_attn, comp_mask, prot_mask);

  // 6) comp_attended = comp_attention @ prot_feat  (NN)
  gemm_wmma_kernel<float, float, true, 0>
      <<<dim3(DD / BN, LC / BM, BATCH), blk, 0, stream>>>(
          comp_attn, (long long)LC * LP, LP,
          prot_feat, (long long)LP * DD, DD,
          (void*)comp_attended, (long long)LC * DD, DD, nullptr, LP);

  // 7) prot_attended = prot_attention @ comp_feat  (NN)
  gemm_wmma_kernel<float, float, true, 0>
      <<<dim3(DD / BN, LP / BM, BATCH), blk, 0, stream>>>(
          prot_attn, (long long)LP * LC, LC,
          comp_feat, (long long)LC * DD, DD,
          (void*)prot_attended, (long long)LP * DD, DD, nullptr, LC);
}
